// BlockRecurrentSwinIRBlock_31293131719187
// MI455X (gfx1250) — matrix operand order repaired and checked by reading, not compiled, not run on hardware
//
#include <hip/hip_runtime.h>
#include <hip/hip_bf16.h>

typedef __bf16 bf16;
typedef __bf16 v16bf __attribute__((ext_vector_type(16)));
typedef __bf16 v8bf  __attribute__((ext_vector_type(8)));
typedef float  v8f   __attribute__((ext_vector_type(8)));

#define WMMA_BF16(a, b, c) \
  __builtin_amdgcn_wmma_f32_16x16x32_bf16(false, (a), false, (b), (short)0, (c), false, false)

// Async global->LDS copy, GVS mode (SGPR64 base + per-lane VGPR32 offset).
// Tracked by ASYNCcnt; 16 bytes per lane per instruction.
#define ASYNC_LD_B128(ldsoff, saddr, voff)                        \
  asm volatile("global_load_async_to_lds_b128 %0, %1, %2"         \
               :: "v"(ldsoff), "v"(voff), "s"(saddr) : "memory")

// ---------------------------------------------------------------------------
// WMMA fragment loaders from LDS (wave32; contiguous k order used for both A and B)
// A (16x32 MxK, 16-bit): lane holds row m=lane%16; K = k_base + hi*16 + e,
//   contiguous (same order as B).
// B (32x16 KxN, 16-bit): lane holds col n=lane%16; K = (lane>=16)*16 + e,
//   contiguous. We store B as Bt[n][k] so this is two adjacent b128 loads.
// ---------------------------------------------------------------------------
__device__ __forceinline__ v16bf frag_a(const bf16* base, int ldk,
                                        int row_base, int k_base, int lane) {
  int m  = row_base + (lane & 15);
  int hi = (lane >> 4) * 16;
  const bf16* rp = base + (size_t)m * ldk + k_base;
  v8bf a0 = *(const v8bf*)(rp + hi);
  v8bf a1 = *(const v8bf*)(rp + hi + 8);
  return __builtin_shufflevector(a0, a1, 0, 1, 2, 3, 4, 5, 6, 7,
                                 8, 9, 10, 11, 12, 13, 14, 15);
}

__device__ __forceinline__ v16bf frag_b(const bf16* base, int ldk,
                                        int n_base, int k_base, int lane) {
  int n = n_base + (lane & 15);
  int k = k_base + (lane >> 4) * 16;
  const v8bf* p = (const v8bf*)(base + (size_t)n * ldk + k);
  v8bf b0 = p[0];
  v8bf b1 = p[1];
  return __builtin_shufflevector(b0, b1, 0, 1, 2, 3, 4, 5, 6, 7,
                                 8, 9, 10, 11, 12, 13, 14, 15);
}

__device__ __forceinline__ void stout(float* p, float v) { *p = v; }
__device__ __forceinline__ void stout(bf16* p, float v)  { *p = (bf16)v; }

// ---------------------------------------------------------------------------
// Setup kernels
// ---------------------------------------------------------------------------
__global__ void cvt_f32_bf16(const float* __restrict__ src,
                             bf16* __restrict__ dst, size_t n) {
  for (size_t i = (size_t)blockIdx.x * blockDim.x + threadIdx.x; i < n;
       i += (size_t)gridDim.x * blockDim.x)
    dst[i] = (bf16)src[i];
}

// src is (K x N) row-major f32; dst is (N x K) row-major bf16 (i.e. W^T)
__global__ void transpose_cvt(const float* __restrict__ src,
                              bf16* __restrict__ dst, int K, int N) {
  size_t total = (size_t)K * N;
  for (size_t i = (size_t)blockIdx.x * blockDim.x + threadIdx.x; i < total;
       i += (size_t)gridDim.x * blockDim.x) {
    int n = (int)(i / K);
    int k = (int)(i % K);
    dst[(size_t)n * K + k] = (bf16)src[(size_t)k * N + n];
  }
}

// biasF[tab][h][n][m] = tab_table[rpi[n*64+m]*8 + h]  (4 * 8 * 64 * 64 f32)
__global__ void bias_expand(const float* __restrict__ t0,
                            const float* __restrict__ t1,
                            const float* __restrict__ t2,
                            const float* __restrict__ t3,
                            const int* __restrict__ rpi,
                            float* __restrict__ biasF) {
  int idx = blockIdx.x * blockDim.x + threadIdx.x; // < 131072
  if (idx >= 4 * 8 * 64 * 64) return;
  int tab = idx >> 15;
  int h   = (idx >> 12) & 7;
  int nm  = idx & 4095;
  const float* t = (tab == 0) ? t0 : (tab == 1) ? t1 : (tab == 2) ? t2 : t3;
  biasF[idx] = t[rpi[nm] * 8 + h];
}

// ---------------------------------------------------------------------------
// GEMM: Y[M x Nld] = X[M x KD] @ W (given as WT[Nld x KD]) + bias.
// 256 threads (8 waves), workgroup tile 128(M) x 128(N); wave -> 32x64
// (2x4 WMMA frags = 8 v_wmma per k-step). Double-buffered LDS staged with
// global_load_async_to_lds_b128 (ASYNCcnt), overlapping HBM with WMMA.
// ---------------------------------------------------------------------------
template <int KD, typename OutT>
__global__ __launch_bounds__(256) void gemm_async(
    const bf16* __restrict__ X, const bf16* __restrict__ WT,
    const float* __restrict__ bias, OutT* __restrict__ Y, int Nld) {
  constexpr int LDS_K = 40;               // 32 + 8 pad (80B rows, 16B aligned)
  constexpr int BUFB  = 128 * LDS_K * 2;  // 10240 bytes per tile buffer
  __shared__ __attribute__((aligned(16))) bf16 As[2][128][LDS_K];
  __shared__ __attribute__((aligned(16))) bf16 Bs[2][128][LDS_K];

  const int t    = threadIdx.x;
  const int lane = t & 31;
  const int wid  = t >> 5;
  const int wm   = wid >> 1;  // 0..3 -> M sub-tile * 32
  const int wn   = wid & 1;   // 0..1 -> N sub-tile * 64
  const size_t mBase = (size_t)blockIdx.x * 128;
  const int    nBase = blockIdx.y * 128;

  // staging coordinates: each thread owns 32B of A and 32B of B per k-step
  const int srow = t >> 1, sseg = (t & 1) * 16;
  const uint32_t asOff = (uint32_t)(uintptr_t)(&As[0][0][0]);
  const uint32_t bsOff = (uint32_t)(uintptr_t)(&Bs[0][0][0]);
  const uint32_t ldsRow = (uint32_t)(srow * LDS_K + sseg) * 2u;

  auto issue = [&](int kt, int buf) {
    const int k0 = kt * 32;
    uint32_t lA = asOff + (uint32_t)buf * BUFB + ldsRow;
    uint32_t gA = (uint32_t)(((mBase + srow) * KD + k0 + sseg) * 2);
    ASYNC_LD_B128(lA,       X, gA);
    ASYNC_LD_B128(lA + 16u, X, gA + 16u);
    uint32_t lB = bsOff + (uint32_t)buf * BUFB + ldsRow;
    uint32_t gB = (uint32_t)(((size_t)(nBase + srow) * KD + k0 + sseg) * 2);
    ASYNC_LD_B128(lB,       WT, gB);
    ASYNC_LD_B128(lB + 16u, WT, gB + 16u);
  };

  v8f c[2][4];
  for (int i = 0; i < 2; i++)
    for (int j = 0; j < 4; j++)
      c[i][j] = (v8f){0, 0, 0, 0, 0, 0, 0, 0};

  constexpr int NK = KD / 32;
  issue(0, 0);
  for (int kt = 0; kt < NK; kt++) {
    const int cur = kt & 1;
    if (kt + 1 < NK) {
      issue(kt + 1, cur ^ 1);  // buffer cur^1 was released by kt-1's barrier
      asm volatile("s_wait_asynccnt 0x4" ::: "memory");  // kt's 4 done (in-order)
    } else {
      asm volatile("s_wait_asynccnt 0x0" ::: "memory");
    }
    __syncthreads();

    const bf16* Ab = &As[cur][0][0];
    const bf16* Bb = &Bs[cur][0][0];
    v16bf a0 = frag_a(Ab, LDS_K, wm * 32, 0, lane);
    v16bf a1 = frag_a(Ab, LDS_K, wm * 32 + 16, 0, lane);
#pragma unroll
    for (int j = 0; j < 4; j++) {
      v16bf bfr = frag_b(Bb, LDS_K, wn * 64 + j * 16, 0, lane);
      c[0][j] = WMMA_BF16(a0, bfr, c[0][j]);
      c[1][j] = WMMA_BF16(a1, bfr, c[1][j]);
    }
    __syncthreads();  // all waves done reading `cur` before it is refilled
  }

  const int hi = lane >> 4, col0 = lane & 15;
  for (int i = 0; i < 2; i++)
    for (int j = 0; j < 4; j++)
      for (int r = 0; r < 8; r++) {
        size_t row = mBase + wm * 32 + i * 16 + r + 8 * hi;
        int    col = nBase + wn * 64 + j * 16 + col0;
        stout(&Y[row * (size_t)Nld + col], c[i][j][r] + bias[col]);
      }
}

// ---------------------------------------------------------------------------
// Attention: one block per (window b, head h, variant vv). 128 threads.
//   vv=0 cross_v: q=qsv(state,j2) k/v=state  tab0 scale=s    -> Av[:,0:256]
//   vv=1 self_v : q=qev(input,j2) k/v=input  tab1 scale=1    -> Av[:,256:512]
//   vv=2 cross_h: q=qeh(input,j3) k/v=input  tab2 scale=s    -> Ah[:,0:256]
//   vv=3 self_h : q=qsh(state,j3) k/v=state  tab3 scale=s*s  -> Ah[:,256:512]
// ---------------------------------------------------------------------------
__global__ __launch_bounds__(128) void attn_kernel(
    const bf16* __restrict__ Ys, const bf16* __restrict__ Ye,
    const float* __restrict__ biasF, bf16* __restrict__ Av,
    bf16* __restrict__ Ah) {
  __shared__ __attribute__((aligned(16))) bf16  qs[64][40];
  __shared__ __attribute__((aligned(16))) bf16  ks2[64][40];
  __shared__ __attribute__((aligned(16))) bf16  vt[32][72];  // vt[d][token]
  __shared__ float scb[64][65];                              // scores f32
  __shared__ __attribute__((aligned(16))) bf16  Pm[64][72];  // softmax probs

  const int b  = blockIdx.x;
  const int h  = blockIdx.y;
  const int vv = blockIdx.z;

  const bf16* src = (vv == 0 || vv == 3) ? Ys : Ye;
  const int   qj  = (vv <= 1) ? 2 : 3;
  const float s   = 0.17677669529663687f; // 1/sqrt(32)
  const float scale = (vv == 1) ? 1.0f : (vv == 3 ? s * s : s);
  bf16* dst = (vv < 2) ? Av : Ah;
  const int halfoff = (vv == 1 || vv == 3) ? 256 : 0;

  const int t = threadIdx.x, lane = t & 31, wid = t >> 5;
  const size_t rowbase = (size_t)b * 64 * 1024;

  { // stage q, k, v(transposed) into LDS
    int n = t >> 1, seg = (t & 1) * 16;
    const v8bf* gq =
        (const v8bf*)(src + rowbase + (size_t)n * 1024 + qj * 256 + h * 32 + seg);
    *(v8bf*)&qs[n][seg]     = gq[0];
    *(v8bf*)&qs[n][seg + 8] = gq[1];
    const v8bf* gk =
        (const v8bf*)(src + rowbase + (size_t)n * 1024 + 0 * 256 + h * 32 + seg);
    *(v8bf*)&ks2[n][seg]     = gk[0];
    *(v8bf*)&ks2[n][seg + 8] = gk[1];
    const v8bf* gv =
        (const v8bf*)(src + rowbase + (size_t)n * 1024 + 1 * 256 + h * 32 + seg);
    v8bf v0 = gv[0], v1 = gv[1];
    for (int i = 0; i < 8; i++) {
      vt[seg + i][n]     = v0[i];
      vt[seg + 8 + i][n] = v1[i];
    }
  }
  __syncthreads();

  // S = q @ k^T : wave wid owns rows [16*wid, 16*wid+16)
  {
    v16bf afr = frag_a(&qs[0][0], 40, 16 * wid, 0, lane);
    v8f   cS[4];
#pragma unroll
    for (int nt = 0; nt < 4; nt++) {
      cS[nt] = (v8f){0, 0, 0, 0, 0, 0, 0, 0};
      v16bf bfr = frag_b(&ks2[0][0], 40, 16 * nt, 0, lane);
      cS[nt] = WMMA_BF16(afr, bfr, cS[nt]);
    }
    const float* bF = biasF + (((size_t)vv * 8 + h) << 12);
    const int hi = lane >> 4, col0 = lane & 15;
    for (int nt = 0; nt < 4; nt++)
      for (int r = 0; r < 8; r++) {
        int row = 16 * wid + r + 8 * hi;
        int col = 16 * nt + col0;
        scb[row][col] = cS[nt][r] * scale + bF[row * 64 + col];
      }
  }
  __syncthreads();

  // softmax over each of 64 rows (threads 0..63)
  if (t < 64) {
    float mx = -1e30f;
    for (int c = 0; c < 64; c++) mx = fmaxf(mx, scb[t][c]);
    float sum = 0.f;
    for (int c = 0; c < 64; c++) {
      float e = __expf(scb[t][c] - mx);
      scb[t][c] = e;
      sum += e;
    }
    float inv = 1.0f / sum;
    for (int c = 0; c < 64; c++) Pm[t][c] = (bf16)(scb[t][c] * inv);
  }
  __syncthreads();

  // O = P @ V : M=64 (wave wid -> 16 rows), N=32 (2 tiles), K=64 (2 steps)
  v8f o[2];
  o[0] = (v8f){0, 0, 0, 0, 0, 0, 0, 0};
  o[1] = (v8f){0, 0, 0, 0, 0, 0, 0, 0};
#pragma unroll
  for (int kk = 0; kk < 64; kk += 32) {
    v16bf afr = frag_a(&Pm[0][0], 72, 16 * wid, kk, lane);
    v16bf b0  = frag_b(&vt[0][0], 72, 0, kk, lane);
    v16bf b1  = frag_b(&vt[0][0], 72, 16, kk, lane);
    o[0] = WMMA_BF16(afr, b0, o[0]);
    o[1] = WMMA_BF16(afr, b1, o[1]);
  }

  const int hi = lane >> 4, col0 = lane & 15;
  for (int j = 0; j < 2; j++)
    for (int r = 0; r < 8; r++) {
      int tok = 16 * wid + r + 8 * hi;
      int d   = 16 * j + col0;
      dst[(size_t)(b * 64 + tok) * 512 + halfoff + h * 32 + d] = (bf16)o[j][r];
    }
}

// ---------------------------------------------------------------------------
extern "C" void kernel_launch(void* const* d_in, const int* in_sizes, int n_in,
                              void* d_out, int out_size, void* d_ws,
                              size_t ws_size, hipStream_t stream) {
  (void)in_sizes; (void)n_in; (void)out_size; (void)ws_size;
  const float* input_x = (const float*)d_in[0];
  const float* state_x = (const float*)d_in[1];
  const float* Ws  = (const float*)d_in[2];
  const float* bs  = (const float*)d_in[3];
  const float* We  = (const float*)d_in[4];
  const float* be  = (const float*)d_in[5];
  const float* Wpv = (const float*)d_in[6];
  const float* bpv = (const float*)d_in[7];
  const float* Wph = (const float*)d_in[8];
  const float* bph = (const float*)d_in[9];
  const float* tcv = (const float*)d_in[10];
  const float* tsv = (const float*)d_in[11];
  const float* tch = (const float*)d_in[12];
  const float* tsh = (const float*)d_in[13];
  const int*   rpi = (const int*)d_in[14];
  float* out = (float*)d_out;

  const size_t M = 131072; // B * N = 2048 * 64
  char* w = (char*)d_ws;
  size_t off = 0;
  auto take = [&](size_t bytes) -> char* {
    char* p = w + off;
    off = (off + bytes + 255) & ~(size_t)255;
    return p;
  };
  bf16* Xe   = (bf16*)take(M * 256 * 2);
  bf16* Xs   = (bf16*)take(M * 256 * 2);
  bf16* WsT  = (bf16*)take(1024 * 256 * 2);
  bf16* WeT  = (bf16*)take(1024 * 256 * 2);
  bf16* WpvT = (bf16*)take(256 * 512 * 2);
  bf16* WphT = (bf16*)take(256 * 512 * 2);
  bf16* Ys   = (bf16*)take(M * 1024 * 2);
  bf16* Ye   = (bf16*)take(M * 1024 * 2);
  bf16* Avb  = (bf16*)take(M * 512 * 2);
  bf16* Ahb  = (bf16*)take(M * 512 * 2);
  float* bF  = (float*)take(4 * 8 * 64 * 64 * 4);

  // 1) conversions / transposes / bias expansion
  cvt_f32_bf16<<<8192, 256, 0, stream>>>(input_x, Xe, M * 256);
  cvt_f32_bf16<<<8192, 256, 0, stream>>>(state_x, Xs, M * 256);
  transpose_cvt<<<1024, 256, 0, stream>>>(Ws, WsT, 256, 1024);
  transpose_cvt<<<1024, 256, 0, stream>>>(We, WeT, 256, 1024);
  transpose_cvt<<<512, 256, 0, stream>>>(Wpv, WpvT, 512, 256);
  transpose_cvt<<<512, 256, 0, stream>>>(Wph, WphT, 512, 256);
  bias_expand<<<512, 256, 0, stream>>>(tcv, tsv, tch, tsh, rpi, bF);

  // 2) QKV projections (WMMA, async double-buffered LDS)
  gemm_async<256, bf16><<<dim3(M / 128, 8), 256, 0, stream>>>(Xs, WsT, bs, Ys, 1024);
  gemm_async<256, bf16><<<dim3(M / 128, 8), 256, 0, stream>>>(Xe, WeT, be, Ye, 1024);

  // 3) 4 attention variants (WMMA QK^T + softmax + WMMA PV)
  attn_kernel<<<dim3(2048, 8, 4), 128, 0, stream>>>(Ys, Ye, bF, Avb, Ahb);

  // 4) output projections (WMMA) straight into d_out (f32)
  gemm_async<512, float><<<dim3(M / 128, 2), 256, 0, stream>>>(Avb, WpvT, bpv, out, 256);
  gemm_async<512, float><<<dim3(M / 128, 2), 256, 0, stream>>>(Ahb, WphT, bph, out + M * 256, 256);
}
